// QADecoder_17145509446238
// MI455X (gfx1250) — matrix operand order repaired and checked by reading, not compiled, not run on hardware
//
#include <hip/hip_runtime.h>
#include <hip/hip_bf16.h>

#define BB 16
#define SS 1024
#define HH 1024
#define MALL (BB * SS)          // 16384 rows total
#define NEGV  (-1e9f)
#define SCALEV 0.03125f          // 1/sqrt(1024)

typedef __attribute__((ext_vector_type(16))) __bf16 bf16x16;
typedef __attribute__((ext_vector_type(8)))  __bf16 bf16x8;
typedef __attribute__((ext_vector_type(8)))  float  f32x8;

// ---------------------------------------------------------------------------
// WMMA fragment loaders (NT GEMM: A is [M,K] row-major, Bt is [N,K] row-major)
// A 16x32 bf16 layout: lanes 0-15 row M=lane hold K=0..15; lanes 16-31
// hold K=16..31, contiguous (matches B).
// B 32x16 layout: lane n holds column n; lanes 0-15 K=0..15, lanes 16-31
// K=16..31, contiguous (same k order as A).
// ---------------------------------------------------------------------------
__device__ __forceinline__ bf16x16 load_fragA(const __bf16* p) {
  bf16x8 lo = *(const bf16x8*)(p);
  bf16x8 hi = *(const bf16x8*)(p + 8);
  return __builtin_shufflevector(lo, hi, 0, 1, 2, 3, 4, 5, 6, 7,
                                 8, 9, 10, 11, 12, 13, 14, 15);
}

__device__ __forceinline__ f32x8 zero8() {
  f32x8 z;
#pragma unroll
  for (int i = 0; i < 8; ++i) z[i] = 0.0f;
  return z;
}

// Each wave computes a 32x64 tile: 2 (M) x 4 (N) accumulators of 16x16.
template <int MT, int NT>
__device__ __forceinline__ void gemm_acc(const __bf16* __restrict__ A, int lda,
                                         const __bf16* __restrict__ Bt, int ldb,
                                         int m0, int n0, int K,
                                         f32x8 acc[MT][NT]) {
  const int lane = threadIdx.x & 31;
  const int r16  = lane & 15;
  const int aoff = (lane & 16) ? 16 : 0;    // A: second K-half group
  const int boff = (lane & 16) ? 16 : 0;   // B: K=16..31 half
  const __bf16* ap[MT];
  const __bf16* bp[NT];
#pragma unroll
  for (int i = 0; i < MT; ++i)
    ap[i] = A + (size_t)(m0 + 16 * i + r16) * lda + aoff;
#pragma unroll
  for (int j = 0; j < NT; ++j)
    bp[j] = Bt + (size_t)(n0 + 16 * j + r16) * ldb + boff;

  for (int k0 = 0; k0 < K; k0 += 32) {
    bf16x16 af[MT], bg[NT];
#pragma unroll
    for (int i = 0; i < MT; ++i) af[i] = load_fragA(ap[i] + k0);
#pragma unroll
    for (int j = 0; j < NT; ++j) bg[j] = *(const bf16x16*)(bp[j] + k0);
#pragma unroll
    for (int i = 0; i < MT; ++i)
#pragma unroll
      for (int j = 0; j < NT; ++j)
        acc[i][j] = __builtin_amdgcn_wmma_f32_16x16x32_bf16(
            false, af[i], false, bg[j], (short)0, acc[i][j], false, false);
  }
}

// C/D 16x16 f32 layout: vgpr r, lanes 0-15 -> (M=r, N=lane); lanes 16-31 ->
// (M=8+r, N=lane-16).
#define TILE_PROLOGUE()                                    \
  const int m0   = (blockIdx.y * 4 + (threadIdx.x >> 5)) * 32; \
  const int n0   = blockIdx.x * 64;                        \
  const int lane = threadIdx.x & 31;                       \
  const int nn   = lane & 15;                              \
  const int mo   = (lane & 16) ? 8 : 0;                    \
  f32x8 acc[2][4];                                         \
  _Pragma("unroll") for (int i = 0; i < 2; ++i)            \
  _Pragma("unroll") for (int j = 0; j < 4; ++j) acc[i][j] = zero8();

// ---------------------------------------------------------------------------
// GEMM kernels. __launch_bounds__(128, 1): 4 waves/block, min 1 wave/SIMD ->
// full VGPR budget, no accumulator spills. All dims divide tile sizes exactly;
// no divergence -> EXEC==all-ones for every WMMA.
// ---------------------------------------------------------------------------

// C = A @ Bt^T + bias, optional tanh, bf16 output.  ACT: 0=none, 1=tanh
template <int ACT>
__global__ __launch_bounds__(128, 1)
void k_gemm_bias_bf16(const __bf16* __restrict__ A,
                      const __bf16* __restrict__ Bt,
                      const float* __restrict__ bias,
                      __bf16* __restrict__ C, int K) {
  TILE_PROLOGUE();
  gemm_acc<2, 4>(A, K, Bt, K, m0, n0, K, acc);
#pragma unroll
  for (int j = 0; j < 4; ++j) {
    const int n = n0 + 16 * j + nn;
    const float bv = bias[n];
#pragma unroll
    for (int i = 0; i < 2; ++i)
#pragma unroll
      for (int rr = 0; rr < 8; ++rr) {
        const int m = m0 + 16 * i + mo + rr;
        float v = acc[i][j][rr] + bv;
        if (ACT == 1) v = tanhf(v);
        C[(size_t)m * HH + n] = (__bf16)v;
      }
  }
}

// V projection with transposed store:  Vt[b][n][s] = (X@Wv + bv)[b*S+s][n]
__global__ __launch_bounds__(128, 1)
void k_gemm_v_bf16t(const __bf16* __restrict__ A,
                    const __bf16* __restrict__ Bt,
                    const float* __restrict__ bias,
                    __bf16* __restrict__ Vt, int K) {
  TILE_PROLOGUE();
  gemm_acc<2, 4>(A, K, Bt, K, m0, n0, K, acc);
#pragma unroll
  for (int j = 0; j < 4; ++j) {
    const int n = n0 + 16 * j + nn;
    const float bv = bias[n];
#pragma unroll
    for (int i = 0; i < 2; ++i)
#pragma unroll
      for (int rr = 0; rr < 8; ++rr) {
        const int m = m0 + 16 * i + mo + rr;      // m = b*SS + s
        const int b = m >> 10;
        const int s = m & (SS - 1);
        Vt[((size_t)(b * HH + n)) * SS + s] = (__bf16)(acc[i][j][rr] + bv);
      }
  }
}

// scores[b] = (Q[b] @ K[b]^T) * scale + addm, XOR bi-attention mask, fp32 out.
__global__ __launch_bounds__(128, 1)
void k_gemm_scores(const __bf16* __restrict__ Q,
                   const __bf16* __restrict__ Kb,
                   const float* __restrict__ mask,
                   const int* __restrict__ divide_pos,
                   float* __restrict__ Sc) {
  const int b = blockIdx.z;
  const __bf16* Aq = Q + (size_t)b * SS * HH;
  const __bf16* Bk = Kb + (size_t)b * SS * HH;
  TILE_PROLOGUE();
  gemm_acc<2, 4>(Aq, HH, Bk, HH, m0, n0, HH, acc);
  const int d = divide_pos[b];
  float* out = Sc + (size_t)b * SS * SS;
#pragma unroll
  for (int j = 0; j < 4; ++j) {
    const int sk = n0 + 16 * j + nn;
    const bool isk = sk < d;
    const float mk = mask[b * SS + sk];
#pragma unroll
    for (int i = 0; i < 2; ++i)
#pragma unroll
      for (int rr = 0; rr < 8; ++rr) {
        const int sq = m0 + 16 * i + mo + rr;
        const bool isq = sq < d;
        const float v = acc[i][j][rr] * SCALEV + (isq ? mk : 0.0f);
        out[(size_t)sq * SS + sk] = (isq != isk) ? v : NEGV;
      }
  }
}

// Row softmax; writes bf16 probabilities in place over the fp32 row
// (all reads happen before the first barrier, writes after the last).
__global__ __launch_bounds__(256)
void k_softmax_p(float* __restrict__ Sc) {
  const size_t row = blockIdx.x;
  float* src = Sc + row * SS;
  __bf16* dst = (__bf16*)src;   // bf16 row stride = 2*SS elements
  __shared__ float red[256];
  const int t = threadIdx.x;
  float v0 = src[t], v1 = src[t + 256], v2 = src[t + 512], v3 = src[t + 768];
  float mx = fmaxf(fmaxf(v0, v1), fmaxf(v2, v3));
  red[t] = mx;
  __syncthreads();
  for (int o = 128; o > 0; o >>= 1) {
    if (t < o) red[t] = fmaxf(red[t], red[t + o]);
    __syncthreads();
  }
  mx = red[0];
  __syncthreads();
  v0 = __expf(v0 - mx); v1 = __expf(v1 - mx);
  v2 = __expf(v2 - mx); v3 = __expf(v3 - mx);
  red[t] = v0 + v1 + v2 + v3;
  __syncthreads();
  for (int o = 128; o > 0; o >>= 1) {
    if (t < o) red[t] += red[t + o];
    __syncthreads();
  }
  const float inv = 1.0f / red[0];
  dst[t] = (__bf16)(v0 * inv);
  dst[t + 256] = (__bf16)(v1 * inv);
  dst[t + 512] = (__bf16)(v2 * inv);
  dst[t + 768] = (__bf16)(v3 * inv);
}

// res[b] = P[b] @ V[b]   (P has lda = 2*SS: bf16 packed inside fp32 rows)
__global__ __launch_bounds__(128, 1)
void k_gemm_pv(const __bf16* __restrict__ P,
               const __bf16* __restrict__ Vt,
               __bf16* __restrict__ res) {
  const int b = blockIdx.z;
  const __bf16* A  = P + (size_t)b * SS * (2 * SS);
  const __bf16* Bt = Vt + (size_t)b * HH * SS;
  TILE_PROLOGUE();
  gemm_acc<2, 4>(A, 2 * SS, Bt, SS, m0, n0, SS, acc);
#pragma unroll
  for (int j = 0; j < 4; ++j) {
    const int n = n0 + 16 * j + nn;
#pragma unroll
    for (int i = 0; i < 2; ++i)
#pragma unroll
      for (int rr = 0; rr < 8; ++rr) {
        const int m = m0 + 16 * i + mo + rr;
        res[((size_t)(b * SS + m)) * HH + n] = (__bf16)acc[i][j][rr];
      }
  }
}

// x = T @ Wd1^T + b_d1 + x ; writes fp32 x (in place) and bf16 copy.
__global__ __launch_bounds__(128, 1)
void k_gemm_d1(const __bf16* __restrict__ A,
               const __bf16* __restrict__ Bt,
               const float* __restrict__ bias,
               float* __restrict__ X, __bf16* __restrict__ Xb,
               int K) {
  TILE_PROLOGUE();
  gemm_acc<2, 4>(A, K, Bt, K, m0, n0, K, acc);
#pragma unroll
  for (int j = 0; j < 4; ++j) {
    const int n = n0 + 16 * j + nn;
    const float bv = bias[n];
#pragma unroll
    for (int i = 0; i < 2; ++i)
#pragma unroll
      for (int rr = 0; rr < 8; ++rr) {
        const int m = m0 + 16 * i + mo + rr;
        const size_t idx = (size_t)m * HH + n;
        const float v = acc[i][j][rr] + bv + X[idx];
        X[idx] = v;
        Xb[idx] = (__bf16)v;
      }
  }
}

// ---------------------------------------------------------------------------
// Small helper kernels
// ---------------------------------------------------------------------------

// Wt[n*HH + k] = (bf16) W[k*HH + n]
__global__ void k_w_transpose_bf16(const float* __restrict__ W,
                                   __bf16* __restrict__ Wt) {
  const int idx = blockIdx.x * 256 + threadIdx.x;
  const int n = idx >> 10;
  const int k = idx & (HH - 1);
  Wt[idx] = (__bf16)W[(size_t)k * HH + n];
}

__global__ void k_f2bf(const float* __restrict__ src, __bf16* __restrict__ dst) {
  const size_t i = (size_t)blockIdx.x * 256 + threadIdx.x;
  dst[i] = (__bf16)src[i];
}

// qa_out[m,0:2], ner_out[m,0:9]; one wave per row, shuffle reduction.
__global__ __launch_bounds__(32)
void k_heads(const float* __restrict__ X,
             const float* __restrict__ wqa, const float* __restrict__ bqa,
             const float* __restrict__ wner, const float* __restrict__ bner,
             float* __restrict__ out) {
  const int row = blockIdx.x;
  const int lane = threadIdx.x;
  const float* x = X + (size_t)row * HH;
  float acc[11];
#pragma unroll
  for (int c = 0; c < 11; ++c) acc[c] = 0.0f;
  for (int k = lane; k < HH; k += 32) {
    const float xv = x[k];
#pragma unroll
    for (int c = 0; c < 2; ++c) acc[c] += xv * wqa[k * 2 + c];
#pragma unroll
    for (int c = 0; c < 9; ++c) acc[2 + c] += xv * wner[k * 9 + c];
  }
#pragma unroll
  for (int c = 0; c < 11; ++c)
    for (int o = 16; o > 0; o >>= 1) acc[c] += __shfl_xor(acc[c], o, 32);
  if (lane == 0) {
#pragma unroll
    for (int c = 0; c < 2; ++c) out[(size_t)row * 2 + c] = acc[c] + bqa[c];
    float* ner = out + (size_t)MALL * 2;
#pragma unroll
    for (int c = 0; c < 9; ++c) ner[(size_t)row * 9 + c] = acc[2 + c] + bner[c];
  }
}

// ---------------------------------------------------------------------------
extern "C" void kernel_launch(void* const* d_in, const int* in_sizes, int n_in,
                              void* d_out, int out_size, void* d_ws, size_t ws_size,
                              hipStream_t stream) {
  const float* input      = (const float*)d_in[0];
  const float* mask       = (const float*)d_in[1];
  const int*   divide_pos = (const int*)d_in[2];
  const float* w_q  = (const float*)d_in[3];  const float* b_q  = (const float*)d_in[4];
  const float* w_k  = (const float*)d_in[5];  const float* b_k  = (const float*)d_in[6];
  const float* w_v  = (const float*)d_in[7];  const float* b_v  = (const float*)d_in[8];
  const float* w_d0 = (const float*)d_in[9];  const float* b_d0 = (const float*)d_in[10];
  const float* w_d1 = (const float*)d_in[11]; const float* b_d1 = (const float*)d_in[12];
  const float* w_qa = (const float*)d_in[13]; const float* b_qa = (const float*)d_in[14];
  const float* w_ner= (const float*)d_in[15]; const float* b_ner= (const float*)d_in[16];

  char* base = (char*)d_ws;
  size_t off = 0;
  auto bump = [&](size_t bytes) -> void* {
    void* p = base + off;
    off += (bytes + 255) & ~(size_t)255;
    return p;
  };
  const size_t wbytes  = (size_t)HH * HH * sizeof(__bf16);        // 2 MB
  const size_t xbbytes = (size_t)MALL * HH * sizeof(__bf16);      // 32 MB
  __bf16* wtq  = (__bf16*)bump(wbytes);
  __bf16* wtk  = (__bf16*)bump(wbytes);
  __bf16* wtv  = (__bf16*)bump(wbytes);
  __bf16* wtd0 = (__bf16*)bump(wbytes);
  __bf16* wtd1 = (__bf16*)bump(wbytes);
  __bf16* xbf  = (__bf16*)bump(xbbytes);
  __bf16* Qb   = (__bf16*)bump(xbbytes);
  __bf16* Kb2  = (__bf16*)bump(xbbytes);
  __bf16* Vt   = (__bf16*)bump(xbbytes);
  float*  Sc   = (float*)bump((size_t)BB * SS * SS * sizeof(float));   // 64 MB
  float*  xcur = (float*)bump((size_t)MALL * HH * sizeof(float));      // 64 MB
  __bf16* resb = Kb2;  // K dead after scores -> reuse for attention output
  __bf16* tb   = Qb;   // Q dead after scores -> reuse for tanh hidden

  hipMemcpyAsync(xcur, input, (size_t)MALL * HH * sizeof(float),
                 hipMemcpyDeviceToDevice, stream);

  const int tblocks = (HH * HH) / 256;  // 4096
  k_w_transpose_bf16<<<tblocks, 256, 0, stream>>>(w_q,  wtq);
  k_w_transpose_bf16<<<tblocks, 256, 0, stream>>>(w_k,  wtk);
  k_w_transpose_bf16<<<tblocks, 256, 0, stream>>>(w_v,  wtv);
  k_w_transpose_bf16<<<tblocks, 256, 0, stream>>>(w_d0, wtd0);
  k_w_transpose_bf16<<<tblocks, 256, 0, stream>>>(w_d1, wtd1);
  k_f2bf<<<(MALL * HH) / 256, 256, 0, stream>>>(input, xbf);

  const dim3 blk(128);                       // 4 waves, each owns a 32x64 tile
  const dim3 gW(HH / 64, MALL / 128);        // 16 x 128 blocks
  const dim3 gS(SS / 64, SS / 128, BB);      // 16 x 8 x 16 blocks

  for (int hop = 0; hop < 3; ++hop) {
    k_gemm_bias_bf16<0><<<gW, blk, 0, stream>>>(xbf, wtq, b_q, Qb, HH);
    k_gemm_bias_bf16<0><<<gW, blk, 0, stream>>>(xbf, wtk, b_k, Kb2, HH);
    k_gemm_v_bf16t<<<gW, blk, 0, stream>>>(xbf, wtv, b_v, Vt, HH);
    k_gemm_scores<<<gS, blk, 0, stream>>>(Qb, Kb2, mask, divide_pos, Sc);
    k_softmax_p<<<MALL, 256, 0, stream>>>(Sc);
    k_gemm_pv<<<gS, blk, 0, stream>>>((const __bf16*)Sc, Vt, resb);
    k_gemm_bias_bf16<1><<<gW, blk, 0, stream>>>(resb, wtd0, b_d0, tb, HH);
    k_gemm_d1<<<gW, blk, 0, stream>>>(tb, wtd1, b_d1, xcur, xbf, HH);
  }

  k_heads<<<MALL, 32, 0, stream>>>(xcur, w_qa, b_qa, w_ner, b_ner, (float*)d_out);
}
